// MultiHeadAttention_58402965291570
// MI455X (gfx1250) — matrix operand order repaired and checked by reading, not compiled, not run on hardware
//
#include <hip/hip_runtime.h>

typedef __bf16 bf16_t;
typedef __attribute__((ext_vector_type(16))) __bf16 v16bf;
typedef __attribute__((ext_vector_type(8)))  __bf16 v8bf;
typedef __attribute__((ext_vector_type(8)))  float  v8f;

constexpr int Bc = 4, Sc = 1024, Ec = 1024, Hc = 16, HDc = 64;
constexpr int LPAD = 8;                 // LDS row padding (floats) -> stride 1032
constexpr size_t MEG = 1048576;         // 1M elements

// ---------------- WMMA fragment helpers (wave32, 16x16x32 bf16) ----------------
// A-matrix 16x32 bf16: lanes 0-15 row m=lane, K = {0..7, 16..23};
//                      lanes 16-31 row m=lane-16, K = {8..15, 24..31}.
__device__ inline v16bf frag_a_bf16(const bf16_t* base, int ld) {
  int lane = threadIdx.x & 31;
  const bf16_t* p = base + (size_t)(lane & 15) * ld + ((lane >> 4) << 3);
  v8bf lo = *(const v8bf*)p;
  v8bf hi = *(const v8bf*)(p + 16);
  return __builtin_shufflevector(lo, hi, 0,1,2,3,4,5,6,7,8,9,10,11,12,13,14,15);
}
// A fragment sourced from fp32 (LDS), converted to bf16 on the fly.
__device__ inline v16bf frag_a_f32(const float* base, int ld) {
  int lane = threadIdx.x & 31;
  const float* p = base + (size_t)(lane & 15) * ld + ((lane >> 4) << 3);
  v16bf a;
#pragma unroll
  for (int i = 0; i < 8; ++i) { a[i] = (bf16_t)p[i]; a[i + 8] = (bf16_t)p[i + 16]; }
  return a;
}
// B-matrix 32x16 bf16: column n = lane&15; lanes 0-15 hold K={0..7,16..23}, lanes 16-31 K={8..15,24..31}.
// element (k, n) lives at base[n*ld + k]  (column-K contiguous).
__device__ inline v16bf frag_b_bf16(const bf16_t* base, int ld) {
  int lane = threadIdx.x & 31;
  const bf16_t* p = base + (size_t)(lane & 15) * ld + ((lane >> 4) << 3);
  v8bf lo = *(const v8bf*)p;
  v8bf hi = *(const v8bf*)(p + 16);
  return __builtin_shufflevector(lo, hi, 0,1,2,3,4,5,6,7,8,9,10,11,12,13,14,15);
}
__device__ inline v8f wmma_bf16(v16bf a, v16bf b, v8f c) {
  return __builtin_amdgcn_wmma_f32_16x16x32_bf16(false, a, false, b, (short)0, c,
                                                 false, false);
}

// ---------------- kernel 1: convert activations, transpose weights ----------------
__global__ __launch_bounds__(256) void prep_kernel(
    const float* __restrict__ q, const float* __restrict__ k, const float* __restrict__ v,
    const float* __restrict__ Wq, const float* __restrict__ Wk, const float* __restrict__ Wv,
    const float* __restrict__ Wfc,
    bf16_t* __restrict__ Xq, bf16_t* __restrict__ Xk, bf16_t* __restrict__ Xv,
    bf16_t* __restrict__ Wqt, bf16_t* __restrict__ Wkt, bf16_t* __restrict__ Wvt,
    bf16_t* __restrict__ Wfcb)
{
  const size_t total = 16 * MEG;
  for (size_t idx = (size_t)blockIdx.x * blockDim.x + threadIdx.x; idx < total;
       idx += (size_t)gridDim.x * blockDim.x) {
    if (idx < 4 * MEG) {
      Xq[idx] = (bf16_t)q[idx];
    } else if (idx < 8 * MEG) {
      size_t i = idx - 4 * MEG; Xk[i] = (bf16_t)k[i];
    } else if (idx < 12 * MEG) {
      size_t i = idx - 8 * MEG; Xv[i] = (bf16_t)v[i];
    } else if (idx < 15 * MEG) {
      // W[h,e,d] -> Wt[j=h*64+d, e]  (column j contiguous over e)
      size_t i = idx - 12 * MEG;
      int which = (int)(i / MEG);
      size_t r = i % MEG;
      int j = (int)(r >> 10), e = (int)(r & 1023);
      const float* W = (which == 0) ? Wq : (which == 1) ? Wk : Wv;
      float val = W[(size_t)(j >> 6) * (Ec * HDc) + (size_t)e * HDc + (j & 63)];
      bf16_t* Wt = (which == 0) ? Wqt : (which == 1) ? Wkt : Wvt;
      Wt[r] = (bf16_t)val;
    } else {
      size_t r = idx - 15 * MEG;
      Wfcb[r] = (bf16_t)Wfc[r];  // Wfc already [j, e] row-major
    }
  }
}

// ---------------- kernel 2: Q/K/V head projections (bf16 WMMA GEMM) ----------------
// [4096 x 1024] x [1024 x 1024]; wave computes 16x64 (A-frag reused over 4 B-frags),
// block = 8 waves -> 32 x 256 tile region.
__global__ __launch_bounds__(256) void proj_kernel(
    const bf16_t* __restrict__ Xq, const bf16_t* __restrict__ Xk, const bf16_t* __restrict__ Xv,
    const bf16_t* __restrict__ Wqt, const bf16_t* __restrict__ Wkt, const bf16_t* __restrict__ Wvt,
    bf16_t* __restrict__ Qp, bf16_t* __restrict__ Kp, bf16_t* __restrict__ Vt)
{
  const int z = blockIdx.z;
  const bf16_t* X  = (z == 0) ? Xq  : (z == 1) ? Xk  : Xv;
  const bf16_t* Wt = (z == 0) ? Wqt : (z == 1) ? Wkt : Wvt;
  const int w = threadIdx.x >> 5, lane = threadIdx.x & 31;
  const int row0 = blockIdx.x * 32 + (w >> 2) * 16;
  const int colg = blockIdx.y * 256 + (w & 3) * 64;

  v8f acc[4] = {};
  for (int e0 = 0; e0 < Ec; e0 += 32) {
    if (e0 + 128 < Ec) {  // gfx1250 global_prefetch_b8 hint a few k-steps ahead
      __builtin_prefetch(X + (size_t)row0 * Ec + e0 + 128, 0, 1);
      __builtin_prefetch(Wt + (size_t)colg * Ec + e0 + 128, 0, 1);
    }
    v16bf a = frag_a_bf16(X + (size_t)row0 * Ec + e0, Ec);
#pragma unroll
    for (int c = 0; c < 4; ++c) {
      v16bf b = frag_b_bf16(Wt + (size_t)(colg + 16 * c) * Ec + e0, Ec);
      acc[c] = wmma_bf16(a, b, acc[c]);
    }
  }
  const int hi = lane >> 4, n = lane & 15;
#pragma unroll
  for (int c = 0; c < 4; ++c) {
    const int j = colg + 16 * c + n, h = j >> 6, d = j & 63;
#pragma unroll
    for (int r = 0; r < 8; ++r) {
      int nr = row0 + hi * 8 + r;
      int bb = nr >> 10, s = nr & 1023;
      size_t bh = (size_t)(bb * Hc + h);
      bf16_t val = (bf16_t)acc[c][r];
      if (z == 0)      Qp[(bh * Sc + s) * HDc + d] = val;    // [B,H,S,64]
      else if (z == 1) Kp[(bh * Sc + s) * HDc + d] = val;    // [B,H,S,64]
      else             Vt[(bh * HDc + d) * Sc + s] = val;    // [B,H,64,S] (transposed)
    }
  }
}

// ---------------- kernel 3: fused logits -> softmax -> sim_cat + score ----------------
// One block = one (b,h) and 32 query rows; padded 32x(1024+8) logit tile in 132KB LDS.
__global__ __launch_bounds__(256) void attn_kernel(
    const bf16_t* __restrict__ Qp, const bf16_t* __restrict__ Kp,
    const bf16_t* __restrict__ Vt, const float* __restrict__ mask,
    float* __restrict__ simout, bf16_t* __restrict__ score)
{
  constexpr int LSTR = Sc + LPAD;       // 1032-float LDS row stride
  __shared__ float lds[32 * LSTR];      // 132096 bytes (WGP LDS = 320KB)
  const int bb = blockIdx.z, h = blockIdx.y;
  const int row_base = blockIdx.x * 32;
  const int w = threadIdx.x >> 5, lane = threadIdx.x & 31;
  const size_t bh = (size_t)bb * Hc + h;
  const bf16_t* Q = Qp + bh * Sc * HDc;
  const bf16_t* K = Kp + bh * Sc * HDc;
  const bf16_t* V = Vt + bh * HDc * Sc;
  const int strip = w >> 2;
  const int hi = lane >> 4, n = lane & 15;

  // -------- phase 1: logits = Q K^T / 64 + mask  (WMMA, K-dim = HD = 64) --------
  v16bf a0 = frag_a_bf16(Q + (size_t)(row_base + strip * 16) * HDc + 0,  HDc);
  v16bf a1 = frag_a_bf16(Q + (size_t)(row_base + strip * 16) * HDc + 32, HDc);
  for (int i = 0; i < 16; ++i) {
    int col0 = ((w & 3) + 4 * i) * 16;
    if (i + 2 < 16)
      __builtin_prefetch(K + (size_t)(col0 + 128) * HDc, 0, 1);
    v16bf b0 = frag_b_bf16(K + (size_t)col0 * HDc + 0,  HDc);
    v16bf b1 = frag_b_bf16(K + (size_t)col0 * HDc + 32, HDc);
    v8f acc = {};
    acc = wmma_bf16(a0, b0, acc);
    acc = wmma_bf16(a1, b1, acc);
#pragma unroll
    for (int r = 0; r < 8; ++r) {
      int rr = strip * 16 + hi * 8 + r;          // local row 0..31
      int s = row_base + rr, t = col0 + n;
      lds[rr * LSTR + t] =
          acc[r] * (1.0f / 64.0f) + mask[((size_t)bb * Sc + s) * Sc + t];
    }
  }
  __syncthreads();

  // -------- phase 2: row softmax (wave32 shuffle reduce), emit sim_cat --------
  for (int rr = w * 4; rr < w * 4 + 4; ++rr) {
    float* row = lds + (size_t)rr * LSTR;
    float mx = -3.4e38f;
#pragma unroll
    for (int i = 0; i < 32; ++i) mx = fmaxf(mx, row[lane + 32 * i]);
#pragma unroll
    for (int off = 16; off > 0; off >>= 1) mx = fmaxf(mx, __shfl_xor(mx, off, 32));
    float sum = 0.f;
#pragma unroll
    for (int i = 0; i < 32; ++i) {
      float e = __expf(row[lane + 32 * i] - mx);
      row[lane + 32 * i] = e;
      sum += e;
    }
#pragma unroll
    for (int off = 16; off > 0; off >>= 1) sum += __shfl_xor(sum, off, 32);
    float inv = 1.0f / sum;
    int s = row_base + rr;
    float* out = simout + ((size_t)bb * Sc + s) * (size_t)(Hc * Sc) + (size_t)h * Sc;
#pragma unroll
    for (int i = 0; i < 32; ++i) {
      float p = row[lane + 32 * i] * inv;
      row[lane + 32 * i] = p;
      out[lane + 32 * i] = p;     // sim_cat[b, s, h*S + t]
    }
  }
  __syncthreads();

  // -------- phase 3: score = sim @ V   ([32x1024] x [1024x64], WMMA) --------
  const int d0 = (w & 3) * 16;
  v8f acc = {};
  for (int t0 = 0; t0 < Sc; t0 += 32) {
    if (t0 + 128 < Sc)
      __builtin_prefetch(V + (size_t)d0 * Sc + t0 + 128, 0, 1);
    v16bf a = frag_a_f32(lds + (size_t)(strip * 16) * LSTR + t0, LSTR);
    v16bf b = frag_b_bf16(V + (size_t)d0 * Sc + t0, Sc);
    acc = wmma_bf16(a, b, acc);
  }
#pragma unroll
  for (int r = 0; r < 8; ++r) {
    int s = row_base + strip * 16 + hi * 8 + r;
    score[((size_t)bb * Sc + s) * Ec + h * HDc + d0 + n] = (bf16_t)acc[r];
  }
}

// ---------------- kernel 4: x = relu(score @ Wfc^T + bfc) ----------------
// wave computes 16x64 (A-frag reused over 4 B-frags); block = 32 x 256 region.
__global__ __launch_bounds__(256) void fc_kernel(
    const bf16_t* __restrict__ score, const bf16_t* __restrict__ Wfcb,
    const float* __restrict__ bfc, float* __restrict__ out)
{
  const int w = threadIdx.x >> 5, lane = threadIdx.x & 31;
  const int row0 = blockIdx.x * 32 + (w >> 2) * 16;
  const int colg = blockIdx.y * 256 + (w & 3) * 64;
  v8f acc[4] = {};
  for (int e0 = 0; e0 < Ec; e0 += 32) {
    if (e0 + 128 < Ec) {
      __builtin_prefetch(score + (size_t)row0 * Ec + e0 + 128, 0, 1);
      __builtin_prefetch(Wfcb + (size_t)colg * Ec + e0 + 128, 0, 1);
    }
    v16bf a = frag_a_bf16(score + (size_t)row0 * Ec + e0, Ec);
#pragma unroll
    for (int c = 0; c < 4; ++c) {
      v16bf b = frag_b_bf16(Wfcb + (size_t)(colg + 16 * c) * Ec + e0, Ec);
      acc[c] = wmma_bf16(a, b, acc[c]);
    }
  }
  const int hi = lane >> 4, n = lane & 15;
#pragma unroll
  for (int c = 0; c < 4; ++c) {
    const int j = colg + 16 * c + n;
    const float bias = bfc[j];
#pragma unroll
    for (int r = 0; r < 8; ++r) {
      int nr = row0 + hi * 8 + r;
      out[(size_t)nr * Ec + j] = fmaxf(acc[c][r] + bias, 0.0f);
    }
  }
}

extern "C" void kernel_launch(void* const* d_in, const int* in_sizes, int n_in,
                              void* d_out, int out_size, void* d_ws, size_t ws_size,
                              hipStream_t stream) {
  const float* query = (const float*)d_in[0];
  const float* key_  = (const float*)d_in[1];
  const float* value = (const float*)d_in[2];
  const float* mask  = (const float*)d_in[3];
  const float* Wq    = (const float*)d_in[4];
  const float* Wk    = (const float*)d_in[5];
  const float* Wv    = (const float*)d_in[6];
  const float* Wfc   = (const float*)d_in[7];
  const float* bfc   = (const float*)d_in[8];
  float* out = (float*)d_out;

  // workspace layout (64 MB total, all bf16 except nothing fp32 now)
  bf16_t* ws   = (bf16_t*)d_ws;
  bf16_t* Xq   = ws;                 // 4M bf16
  bf16_t* Xk   = ws + 4 * MEG;       // 4M
  bf16_t* Xv   = ws + 8 * MEG;       // 4M
  bf16_t* Wqt  = ws + 12 * MEG;      // 1M
  bf16_t* Wkt  = ws + 13 * MEG;      // 1M
  bf16_t* Wvt  = ws + 14 * MEG;      // 1M
  bf16_t* Wfcb = ws + 15 * MEG;      // 1M
  bf16_t* Qp   = ws + 16 * MEG;      // 4M  [B,H,S,64]
  bf16_t* Kp   = ws + 20 * MEG;      // 4M  [B,H,S,64]
  bf16_t* Vt   = ws + 24 * MEG;      // 4M  [B,H,64,S]
  bf16_t* scoreb = ws + 28 * MEG;    // 4M  [B,S,E] bf16

  float* x_out   = out;              // [B,S,E]   = 4M floats
  float* sim_out = out + 4 * MEG;    // [B,S,H*S] = 64M floats

  prep_kernel<<<4096, 256, 0, stream>>>(query, key_, value, Wq, Wk, Wv, Wfc,
                                        Xq, Xk, Xv, Wqt, Wkt, Wvt, Wfcb);
  proj_kernel<<<dim3(128, 4, 3), 256, 0, stream>>>(Xq, Xk, Xv, Wqt, Wkt, Wvt,
                                                   Qp, Kp, Vt);
  attn_kernel<<<dim3(Sc / 32, Hc, Bc), 256, 0, stream>>>(Qp, Kp, Vt, mask,
                                                         sim_out, scoreb);
  fc_kernel<<<dim3(128, 4), 256, 0, stream>>>(scoreb, Wfcb, bfc, x_out);
}
